// ProbSparseAttention_52785148068649
// MI455X (gfx1250) — matrix operand order repaired and checked by reading, not compiled, not run on hardware
//
#include <hip/hip_runtime.h>
#include <hip/hip_bf16.h>
#include <math.h>

typedef float v8f  __attribute__((ext_vector_type(8)));
typedef __bf16 v16bf __attribute__((ext_vector_type(16)));
typedef __bf16 v8bf  __attribute__((ext_vector_type(8)));

#define Bq    8
#define Lq    2048
#define DMq   512
#define Hq    8
#define DHq   64
#define SKq   24
#define NTOPq 24

#define NEG_BIG -3.402823466e38f

// ---------------------------------------------------------------------------
// LayerNorm: one block per row of DMq elements. eps = 1e-8 (matches reference).
// ---------------------------------------------------------------------------
__global__ void __launch_bounds__(256)
ps_ln_kernel(const float* __restrict__ x, const float* __restrict__ w,
             const float* __restrict__ b, float* __restrict__ y) {
  const int row = blockIdx.x;
  const float* xr = x + (size_t)row * DMq;
  float* yr = y + (size_t)row * DMq;
  __shared__ float r1[256], r2[256];
  float s = 0.f, s2 = 0.f;
  for (int d = threadIdx.x; d < DMq; d += blockDim.x) {
    float t = xr[d]; s += t; s2 += t * t;
  }
  r1[threadIdx.x] = s; r2[threadIdx.x] = s2;
  __syncthreads();
  for (int st = 128; st; st >>= 1) {
    if (threadIdx.x < st) {
      r1[threadIdx.x] += r1[threadIdx.x + st];
      r2[threadIdx.x] += r2[threadIdx.x + st];
    }
    __syncthreads();
  }
  const float mean = r1[0] * (1.0f / DMq);
  const float var  = r2[0] * (1.0f / DMq) - mean * mean;
  const float inv  = rsqrtf(var + 1e-8f);
  for (int d = threadIdx.x; d < DMq; d += blockDim.x)
    yr[d] = w[d] * (xr[d] - mean) * inv + b[d];
}

// ---------------------------------------------------------------------------
// fp32 -> bf16 hi/lo split (error-compensated precision): x = hi + lo with
// hi = bf16(x), lo = bf16(x - hi).  Row-major planes for activations.
// ---------------------------------------------------------------------------
__global__ void __launch_bounds__(256)
ps_split_bf16(const float* __restrict__ x, __bf16* __restrict__ hi,
              __bf16* __restrict__ lo, int n) {
  const int i = blockIdx.x * blockDim.x + threadIdx.x;
  if (i >= n) return;
  const float v = x[i];
  const __bf16 h = (__bf16)v;
  hi[i] = h;
  lo[i] = (__bf16)(v - (float)h);
}

// Weight split with transpose: out[n][k] = split(W[k][n]).  Transposed planes
// make each WMMA B-fragment a single contiguous 32B load per plane.
__global__ void __launch_bounds__(256)
ps_split_bf16_wT(const float* __restrict__ w, __bf16* __restrict__ hiT,
                 __bf16* __restrict__ loT) {
  const int idx = blockIdx.x * blockDim.x + threadIdx.x;   // over K*N
  if (idx >= DMq * DMq) return;
  const int kk = idx / DMq, nn = idx - kk * DMq;           // coalesced read
  const float v = w[idx];
  const __bf16 h = (__bf16)v;
  hiT[(size_t)nn * DMq + kk] = h;
  loT[(size_t)nn * DMq + kk] = (__bf16)(v - (float)h);
}

// ---------------------------------------------------------------------------
// Dense GEMM on the bf16 matrix pipe with hi/lo error compensation:
//   C = A @ W + bias,  A ~ Ahi+Alo, W ~ Whi+Wlo (transposed planes),
//   acc += Ahi*Whi + Ahi*Wlo + Alo*Whi   (3x v_wmma_f32_16x16x32_bf16 / K=32)
// One wave computes a 16x64 C strip (4 tiles) to amortize A-fragment loads.
// Fragment layouts (C/D per CDNA5 ISA 7.12.2):
//   16-bit A 16x32: lane group h, element e -> K = 16h+e (this code's contiguous order; ISA table differs)
//   16-bit B 32x16: lane group h, element e -> K = 16h+e (same order as A), column = lane%16
//   32-bit C/D 16x16: VGPR r -> row r+8h, lane%16 -> column
// ---------------------------------------------------------------------------
__global__ void __launch_bounds__(256)
ps_gemm_wmma_bf16x2(const __bf16* __restrict__ Ahi, const __bf16* __restrict__ Alo,
                    const __bf16* __restrict__ WThi, const __bf16* __restrict__ WTlo,
                    const float* __restrict__ bias, float* __restrict__ C,
                    int Mrows, int K, int N) {
  const int wave = blockIdx.x * (blockDim.x >> 5) + (threadIdx.x >> 5);
  const int lane = threadIdx.x & 31;
  const int stripsN = N >> 6;                       // 64-wide output strips
  const int total = (Mrows >> 4) * stripsN;
  if (wave >= total) return;                        // wave-uniform: EXEC all-1s
  const int tm = wave / stripsN;
  const int tn = wave - tm * stripsN;
  const int row0 = tm << 4, col0 = tn << 6;
  const int half = lane >> 4, l16 = lane & 15;

  v8f acc[4];
#pragma unroll
  for (int t = 0; t < 4; ++t) {
    const float bc = bias[col0 + 16 * t + l16];
#pragma unroll
    for (int r = 0; r < 8; ++r) acc[t][r] = bc;
  }

  const __bf16* aH = Ahi + (size_t)(row0 + l16) * K;
  const __bf16* aL = Alo + (size_t)(row0 + l16) * K;

  for (int k0 = 0; k0 < K; k0 += 32) {
    const v8bf h0 = *(const v8bf*)(aH + k0 + 16 * half);
    const v8bf h1 = *(const v8bf*)(aH + k0 + 16 * half + 8);
    const v8bf l0 = *(const v8bf*)(aL + k0 + 16 * half);
    const v8bf l1 = *(const v8bf*)(aL + k0 + 16 * half + 8);
    v16bf ahi, alo;
#pragma unroll
    for (int e = 0; e < 8; ++e) {
      ahi[e] = h0[e]; ahi[8 + e] = h1[e];
      alo[e] = l0[e]; alo[8 + e] = l1[e];
    }
#pragma unroll
    for (int t = 0; t < 4; ++t) {
      const size_t woff = (size_t)(col0 + 16 * t + l16) * K + k0 + 16 * half;
      const v16bf bhi = *(const v16bf*)(WThi + woff);
      const v16bf blo = *(const v16bf*)(WTlo + woff);
      acc[t] = __builtin_amdgcn_wmma_f32_16x16x32_bf16(
          false, ahi, false, bhi, (short)0, acc[t], false, false);
      acc[t] = __builtin_amdgcn_wmma_f32_16x16x32_bf16(
          false, ahi, false, blo, (short)0, acc[t], false, false);
      acc[t] = __builtin_amdgcn_wmma_f32_16x16x32_bf16(
          false, alo, false, bhi, (short)0, acc[t], false, false);
    }
  }

#pragma unroll
  for (int t = 0; t < 4; ++t) {
    float* Crow = C + (size_t)(row0 + half * 8) * N + (col0 + 16 * t + l16);
#pragma unroll
    for (int r = 0; r < 8; ++r) Crow[(size_t)r * N] = acc[t][r];
  }
}

// ---------------------------------------------------------------------------
// Sampled scoring: M[b,h,i] = max_s(QK_s) - sum_s(QK_s)/L over SK sampled keys.
// One wave32 per (b,h,i); lane covers dh = {lane, lane+32} of DH=64.
// ---------------------------------------------------------------------------
__global__ void __launch_bounds__(256)
ps_qks_kernel(const float* __restrict__ q, const float* __restrict__ k,
              const int* __restrict__ idx, float* __restrict__ M) {
  const int wave = blockIdx.x * (blockDim.x >> 5) + (threadIdx.x >> 5);
  const int lane = threadIdx.x & 31;
  const int total = Bq * Hq * Lq;
  if (wave >= total) return;
  const int i  = wave % Lq;
  const int bh = wave / Lq;
  const int h  = bh % Hq;
  const int b  = bh / Hq;

  const float* qrow = q + (size_t)(b * Lq + i) * DMq + h * DHq;
  const float q0 = qrow[lane];
  const float q1 = qrow[lane + 32];

  float mx = NEG_BIG, sm = 0.f;
  for (int s = 0; s < SKq; ++s) {
    const int j = idx[i * SKq + s];
    const float* krow = k + (size_t)(b * Lq + j) * DMq + h * DHq;
    float p = q0 * krow[lane] + q1 * krow[lane + 32];
    for (int off = 16; off; off >>= 1) p += __shfl_xor(p, off, 32);
    mx = fmaxf(mx, p);
    sm += p;
  }
  if (lane == 0) M[wave] = mx - sm * (1.0f / (float)Lq);
}

// ---------------------------------------------------------------------------
// Top-NTOP selection per (b,h): iterative argmax, lowest-index tie-break.
// ---------------------------------------------------------------------------
__global__ void __launch_bounds__(256)
ps_topk_kernel(const float* __restrict__ M, int* __restrict__ mtop) {
  const int bh = blockIdx.x;
  const float* Mr = M + (size_t)bh * Lq;
  __shared__ float vals[Lq];
  __shared__ float rv[256];
  __shared__ int   ri[256];
  for (int i = threadIdx.x; i < Lq; i += blockDim.x) vals[i] = Mr[i];
  __syncthreads();
  for (int t = 0; t < NTOPq; ++t) {
    float bv = NEG_BIG; int bi = 0x7fffffff;
    for (int i = threadIdx.x; i < Lq; i += blockDim.x) {
      float v = vals[i];
      if (v > bv || (v == bv && i < bi)) { bv = v; bi = i; }
    }
    rv[threadIdx.x] = bv; ri[threadIdx.x] = bi;
    __syncthreads();
    for (int st = 128; st; st >>= 1) {
      if (threadIdx.x < st) {
        float ov = rv[threadIdx.x + st]; int oi = ri[threadIdx.x + st];
        if (ov > rv[threadIdx.x] ||
            (ov == rv[threadIdx.x] && oi < ri[threadIdx.x])) {
          rv[threadIdx.x] = ov; ri[threadIdx.x] = oi;
        }
      }
      __syncthreads();
    }
    if (threadIdx.x == 0) {
      mtop[bh * NTOPq + t] = ri[0];
      vals[ri[0]] = NEG_BIG;
    }
    __syncthreads();
  }
}

// ---------------------------------------------------------------------------
// v_avg = cumsum(v, L) / (cumsum(pad, L) + 1e-12); one thread per (b, d)
// column, coalesced across threads for each step of the serial scan.
// ---------------------------------------------------------------------------
__global__ void __launch_bounds__(256)
ps_cumavg_kernel(const float* __restrict__ v, const float* __restrict__ pad,
                 float* __restrict__ outp) {
  const int t = blockIdx.x * blockDim.x + threadIdx.x;
  if (t >= Bq * DMq) return;
  const int b = t / DMq, d = t - b * DMq;
  const float* vb = v + (size_t)b * Lq * DMq + d;
  const float* pb = pad + (size_t)b * Lq;
  float* ob = outp + (size_t)b * Lq * DMq + d;
  float sv = 0.f, sp = 0.f;
  for (int l = 0; l < Lq; ++l) {
    sv += vb[(size_t)l * DMq];
    sp += pb[l];
    ob[(size_t)l * DMq] = sv / (sp + 1e-12f);
  }
}

// ---------------------------------------------------------------------------
// Sparse attention over the NTOP selected queries; one block per (b,h,u).
// Scores masked with -100000.0 exactly like the reference, softmax over full L,
// result scatters into the cumavg output at row m_top[u], cols [h*DH, h*DH+64).
// ---------------------------------------------------------------------------
__global__ void __launch_bounds__(256)
ps_spattn_kernel(const float* __restrict__ q, const float* __restrict__ k,
                 const float* __restrict__ v, const float* __restrict__ pad,
                 const int* __restrict__ mtop, float* __restrict__ outp) {
  const int blk = blockIdx.x;            // (b*H + h)*NTOP + u
  const int u  = blk % NTOPq;
  const int bh = blk / NTOPq;
  const int h  = bh % Hq;
  const int b  = bh / Hq;
  const int i  = mtop[bh * NTOPq + u];
  const float scale = 0.044194173824159216f;   // 1/sqrt(DM=512)

  __shared__ float qrow[DHq];
  __shared__ float sc[Lq];
  __shared__ float red[256];

  if (threadIdx.x < DHq)
    qrow[threadIdx.x] = q[(size_t)(b * Lq + i) * DMq + h * DHq + threadIdx.x];
  __syncthreads();
  const bool rowvalid = (pad[(size_t)b * Lq + i] != 0.f);

  for (int col = threadIdx.x; col < Lq; col += blockDim.x) {
    const float* kr = k + (size_t)(b * Lq + col) * DMq + h * DHq;
    float s = 0.f;
#pragma unroll
    for (int d = 0; d < DHq; ++d) s += qrow[d] * kr[d];
    s *= scale;
    if (!(col <= i && rowvalid)) s = -100000.0f;
    sc[col] = s;
  }
  __syncthreads();

  float m = NEG_BIG;
  for (int col = threadIdx.x; col < Lq; col += blockDim.x) m = fmaxf(m, sc[col]);
  red[threadIdx.x] = m;
  __syncthreads();
  for (int st = 128; st; st >>= 1) {
    if (threadIdx.x < st) red[threadIdx.x] = fmaxf(red[threadIdx.x], red[threadIdx.x + st]);
    __syncthreads();
  }
  const float mglob = red[0];
  __syncthreads();

  float ss = 0.f;
  for (int col = threadIdx.x; col < Lq; col += blockDim.x) {
    float e = expf(sc[col] - mglob);
    sc[col] = e;
    ss += e;
  }
  red[threadIdx.x] = ss;
  __syncthreads();
  for (int st = 128; st; st >>= 1) {
    if (threadIdx.x < st) red[threadIdx.x] += red[threadIdx.x + st];
    __syncthreads();
  }
  const float inv = 1.0f / red[0];
  __syncthreads();

  const int dh  = threadIdx.x & 63;
  const int grp = threadIdx.x >> 6;      // 0..3
  float acc = 0.f;
  for (int col = grp; col < Lq; col += 4)
    acc += sc[col] * v[(size_t)(b * Lq + col) * DMq + h * DHq + dh];
  red[threadIdx.x] = acc;
  __syncthreads();
  if (threadIdx.x < DHq) {
    float uval = red[threadIdx.x] + red[threadIdx.x + 64] +
                 red[threadIdx.x + 128] + red[threadIdx.x + 192];
    outp[(size_t)(b * Lq + i) * DMq + h * DHq + threadIdx.x] = uval * inv;
  }
}

// ---------------------------------------------------------------------------
extern "C" void kernel_launch(void* const* d_in, const int* in_sizes, int n_in,
                              void* d_out, int out_size, void* d_ws, size_t ws_size,
                              hipStream_t stream) {
  const float* queries = (const float*)d_in[0];
  const float* keys    = (const float*)d_in[1];
  const float* values  = (const float*)d_in[2];
  const float* padmask = (const float*)d_in[3];
  const float* Wq = (const float*)d_in[4];
  const float* bqv = (const float*)d_in[5];
  const float* Wk = (const float*)d_in[6];
  const float* bkv = (const float*)d_in[7];
  const float* Wv = (const float*)d_in[8];
  const float* bvv = (const float*)d_in[9];
  const float* Wf = (const float*)d_in[10];
  const float* bfv = (const float*)d_in[11];
  const float* qln_w = (const float*)d_in[12];
  const float* qln_b = (const float*)d_in[13];
  const float* fln_w = (const float*)d_in[14];
  const float* fln_b = (const float*)d_in[15];
  const int* index_sample = (const int*)d_in[16];
  float* out = (float*)d_out;

  const size_t mat  = (size_t)Bq * Lq * DMq;     // 8,388,608 floats
  const size_t wmat = (size_t)DMq * DMq;         // 262,144 floats

  float* ws     = (float*)d_ws;
  float* qn     = ws;                            // LN(queries); reused as fproj
  float* qp     = qn + mat;
  float* kp     = qp + mat;
  float* vp     = kp + mat;
  float* Mbuf   = vp + mat;                      // B*H*L
  int*   mtop   = (int*)(Mbuf + (size_t)Bq * Hq * Lq);
  float* outpre = (float*)(mtop + Bq * Hq * NTOPq);
  // bf16 scratch planes (reused across the sequential GEMMs)
  __bf16* aHi = (__bf16*)(outpre + mat);
  __bf16* aLo = aHi + mat;
  __bf16* wHi = aLo + mat;                       // transposed (N x K)
  __bf16* wLo = wHi + wmat;
  float*  fproj = qn;                            // qn dead after q projection

  const int rows = Bq * Lq;                      // 16384
  const int gemmBlocks  = ((rows / 16) * (DMq / 64)) / 8;   // 8 waves/block
  const int splitBlocks = (int)(mat / 256);
  const int wsplitBlocks = (int)(wmat / 256);

  // 1. LayerNorm(queries)
  ps_ln_kernel<<<rows, 256, 0, stream>>>(queries, qln_w, qln_b, qn);

  // 2. Q projection
  ps_split_bf16<<<splitBlocks, 256, 0, stream>>>(qn, aHi, aLo, (int)mat);
  ps_split_bf16_wT<<<wsplitBlocks, 256, 0, stream>>>(Wq, wHi, wLo);
  ps_gemm_wmma_bf16x2<<<gemmBlocks, 256, 0, stream>>>(aHi, aLo, wHi, wLo, bqv, qp, rows, DMq, DMq);

  // 3. K projection
  ps_split_bf16<<<splitBlocks, 256, 0, stream>>>(keys, aHi, aLo, (int)mat);
  ps_split_bf16_wT<<<wsplitBlocks, 256, 0, stream>>>(Wk, wHi, wLo);
  ps_gemm_wmma_bf16x2<<<gemmBlocks, 256, 0, stream>>>(aHi, aLo, wHi, wLo, bkv, kp, rows, DMq, DMq);

  // 4. V projection
  ps_split_bf16<<<splitBlocks, 256, 0, stream>>>(values, aHi, aLo, (int)mat);
  ps_split_bf16_wT<<<wsplitBlocks, 256, 0, stream>>>(Wv, wHi, wLo);
  ps_gemm_wmma_bf16x2<<<gemmBlocks, 256, 0, stream>>>(aHi, aLo, wHi, wLo, bvv, vp, rows, DMq, DMq);

  // 5. Sampled scoring -> M
  ps_qks_kernel<<<(Bq * Hq * Lq) / 8, 256, 0, stream>>>(qp, kp, index_sample, Mbuf);

  // 6. Top-NTOP per (b,h)
  ps_topk_kernel<<<Bq * Hq, 256, 0, stream>>>(Mbuf, mtop);

  // 7. Cumulative average of V -> default output rows
  ps_cumavg_kernel<<<(Bq * DMq) / 256, 256, 0, stream>>>(vp, padmask, outpre);

  // 8. Sparse attention updates scattered over selected rows
  ps_spattn_kernel<<<Bq * Hq * NTOPq, 256, 0, stream>>>(qp, kp, vp, padmask, mtop, outpre);

  // 9. Output projection + final LayerNorm
  ps_split_bf16<<<splitBlocks, 256, 0, stream>>>(outpre, aHi, aLo, (int)mat);
  ps_split_bf16_wT<<<wsplitBlocks, 256, 0, stream>>>(Wf, wHi, wLo);
  ps_gemm_wmma_bf16x2<<<gemmBlocks, 256, 0, stream>>>(aHi, aLo, wHi, wLo, bfv, fproj, rows, DMq, DMq);
  ps_ln_kernel<<<rows, 256, 0, stream>>>(fproj, fln_w, fln_b, out);
}
